// HRRNSVQ_86431921865286
// MI455X (gfx1250) — matrix operand order repaired and checked by reading, not compiled, not run on hardware
//
#include <hip/hip_runtime.h>
#include <hip/hip_bf16.h>
#include <math.h>

// ---------------------------------------------------------------------------
// HRR noise-substitution VQ for MI455X (gfx1250, wave32, WMMA).
// N=65536 rows, K=4096 codes, D=256 features, fp32 in/out.
//
// GEMM part (x @ C^T, 137 GFLOP) done with split-bf16 (hi+lo) WMMA:
//   x.c ~= xh.ch + xl.ch + xh.cl  (3 independent v_wmma chains per chunk)
// B chunks staged global->LDS with GLOBAL_LOAD_ASYNC_TO_LDS_B128 when the
// toolchain exposes the builtin (ASYNCcnt path), else via VGPR staging.
// ---------------------------------------------------------------------------

#define NN 65536
#define KK 4096
#define DD 256

typedef __attribute__((ext_vector_type(16))) __bf16 v16bf;
typedef __attribute__((ext_vector_type(8)))  float  v8f;
typedef int v4i_vs __attribute__((vector_size(16)));

union BF16x16 { v16bf v; unsigned short s[16]; };

#if __has_builtin(__builtin_amdgcn_global_load_async_to_lds_b128)
#define HAVE_ASYNC_LDS 1
// Builtin expects: (v4i addrspace(1)* gsrc, v4i addrspace(3)* ldst, imm off, imm cpol)
#define GPTR(p) ((__attribute__((address_space(1))) v4i_vs*)(p))
#define LPTR(p) ((__attribute__((address_space(3))) v4i_vs*)(p))
#else
#define HAVE_ASYNC_LDS 0
#endif

__device__ __forceinline__ void wait_async0() {
#if __has_builtin(__builtin_amdgcn_s_wait_asynccnt)
    __builtin_amdgcn_s_wait_asynccnt(0);
#else
    asm volatile("s_wait_asynccnt 0" ::: "memory");
#endif
}

__device__ __forceinline__ unsigned short f2bf(float x) {
    unsigned u = __float_as_uint(x);
    unsigned r = u + 0x7FFFu + ((u >> 16) & 1u);   // round-to-nearest-even
    return (unsigned short)(r >> 16);
}
__device__ __forceinline__ float bf2f(unsigned short h) {
    return __uint_as_float(((unsigned)h) << 16);
}

__device__ __forceinline__ v8f wmma_bf16(v16bf a, v16bf b, v8f c) {
    return __builtin_amdgcn_wmma_f32_16x16x32_bf16(
        /*neg_a=*/false, a, /*neg_b=*/false, b,
        /*c_mod=*/(short)0, c, /*reuse_a=*/false, /*reuse_b=*/false);
}

// ---------------------------------------------------------------------------
// Kernel 1: pack codebook into WMMA-B lane order (bf16 hi/lo planes) and
// compute ||c||^2 per code.
//
// Packed layout (bf16 units):
//   idx = ((ntile*8 + kstep)*2 + plane)*512 + lane*16 + e
//   where code  = ntile*16 + (lane & 15)
//         feat  = kstep*32 + ((lane & 16) ? 16 : 0) + e     (e = 0..15)
// Each lane of a wave reads 16 contiguous bf16 (32B) per (kstep, plane):
// a 32x16 bf16 B operand in contiguous k order (k = 16*(lane/16) + e), same as the A load.
// ---------------------------------------------------------------------------
__global__ void vq_pack_codebook(const float* __restrict__ cb,
                                 unsigned short* __restrict__ packB,
                                 float* __restrict__ cnorm2) {
    int c = blockIdx.x;      // code 0..4095
    int f = threadIdx.x;     // feature 0..255

    float val = cb[(size_t)c * DD + f];
    unsigned short hb = f2bf(val);
    unsigned short lb = f2bf(val - bf2f(hb));

    int col   = c & 15;
    int nt    = c >> 4;
    int kstep = f >> 5;
    int r     = f & 31;
    int lane  = col + ((r & 16) ? 16 : 0);
    int e     = r & 15;
    size_t bidx = ((size_t)(nt * 8 + kstep) * 2) * 512 + (size_t)lane * 16 + e;
    packB[bidx]       = hb;   // hi plane
    packB[bidx + 512] = lb;   // lo plane

    __shared__ float red[256];
    red[f] = val * val;
    __syncthreads();
    for (int s = 128; s > 0; s >>= 1) {
        if (f < s) red[f] += red[f + s];
        __syncthreads();
    }
    if (f == 0) cnorm2[c] = red[0];
}

// ---------------------------------------------------------------------------
// Kernel 2: fused distance GEMM + argmin.
// Grid: 512 blocks x 256 threads (8 waves). Each wave owns 16 rows of x
// (A operand resident in VGPRs as bf16 hi/lo), loops over 256 chunks of
// 16 codes. B chunks (16KB) double-buffered through LDS, shared by 8 waves.
// ---------------------------------------------------------------------------
__global__ void __launch_bounds__(256)
vq_argmin(const float* __restrict__ x,
          const unsigned short* __restrict__ packB,
          const float* __restrict__ cnorm2,
          int* __restrict__ indices) {
    __shared__ uint4 ldsbuf[2048];   // 2 x 16KB double buffer

    const int lane = threadIdx.x & 31;
    const int wave = threadIdx.x >> 5;
    const int rowBase = blockIdx.x * 128 + wave * 16;

    // ---- Load A tile (16 rows x 256 feats) as bf16 hi/lo in WMMA-A order.
    // lane l<16 : row l,    feats {k*32 + 0..15}
    // lane l>=16: row l-16, feats {k*32 + 16..31}
    const float* xr = x + (size_t)(rowBase + (lane & 15)) * DD;
    const int s0 = (lane & 16) ? 16 : 0;
    v16bf Ahi[8], Alo[8];
#pragma unroll
    for (int k = 0; k < 8; ++k) {
        BF16x16 h, l;
#pragma unroll
        for (int part = 0; part < 2; ++part) {
            int f = k * 32 + s0 + part * 8;
            float4 v0 = *(const float4*)(xr + f);
            float4 v1 = *(const float4*)(xr + f + 4);
            float tmp[8] = {v0.x, v0.y, v0.z, v0.w, v1.x, v1.y, v1.z, v1.w};
#pragma unroll
            for (int e = 0; e < 8; ++e) {
                unsigned short hb = f2bf(tmp[e]);
                h.s[part * 8 + e] = hb;
                l.s[part * 8 + e] = f2bf(tmp[e] - bf2f(hb));
            }
        }
        Ahi[k] = h.v;
        Alo[k] = l.v;
    }

    float minv[8];
    int   mini[8];
#pragma unroll
    for (int j = 0; j < 8; ++j) { minv[j] = 3.4e38f; mini[j] = 0; }

    const uint4* gsrc = (const uint4*)packB;   // 1024 uint4 per 16-code chunk
    const int t = threadIdx.x;

    // ---- Preload chunk 0 into buffer 0.
#if HAVE_ASYNC_LDS
    {
        const uint4* g = gsrc + (size_t)t * 4;
        uint4* l = &ldsbuf[t * 4];
        __builtin_amdgcn_global_load_async_to_lds_b128(GPTR(g), LPTR(l),  0, 0);
        __builtin_amdgcn_global_load_async_to_lds_b128(GPTR(g), LPTR(l), 16, 0);
        __builtin_amdgcn_global_load_async_to_lds_b128(GPTR(g), LPTR(l), 32, 0);
        __builtin_amdgcn_global_load_async_to_lds_b128(GPTR(g), LPTR(l), 48, 0);
    }
    wait_async0();
#else
    {
        uint4 r0 = gsrc[t * 4 + 0];
        uint4 r1 = gsrc[t * 4 + 1];
        uint4 r2 = gsrc[t * 4 + 2];
        uint4 r3 = gsrc[t * 4 + 3];
        ldsbuf[t * 4 + 0] = r0;
        ldsbuf[t * 4 + 1] = r1;
        ldsbuf[t * 4 + 2] = r2;
        ldsbuf[t * 4 + 3] = r3;
    }
#endif
    __syncthreads();

    for (int c = 0; c < 256; ++c) {
        const int cur = c & 1;
        const bool more = (c + 1) < 256;

#if HAVE_ASYNC_LDS
        // Kick off async staging of the next chunk right away: the previous
        // barrier guarantees every wave is done reading that buffer.
        if (more) {
            const uint4* g = gsrc + (size_t)(c + 1) * 1024 + (size_t)t * 4;
            uint4* l = &ldsbuf[((c + 1) & 1) * 1024 + t * 4];
            __builtin_amdgcn_global_load_async_to_lds_b128(GPTR(g), LPTR(l),  0, 0);
            __builtin_amdgcn_global_load_async_to_lds_b128(GPTR(g), LPTR(l), 16, 0);
            __builtin_amdgcn_global_load_async_to_lds_b128(GPTR(g), LPTR(l), 32, 0);
            __builtin_amdgcn_global_load_async_to_lds_b128(GPTR(g), LPTR(l), 48, 0);
        }
#else
        uint4 r0, r1, r2, r3;
        if (more) {
            size_t gb = (size_t)(c + 1) * 1024 + (size_t)t * 4;
            r0 = gsrc[gb + 0];
            r1 = gsrc[gb + 1];
            r2 = gsrc[gb + 2];
            r3 = gsrc[gb + 3];
        }
#endif

        // ---- Compute 16x16 dot-product tile for this chunk.
        // Three independent accumulator chains (depth 8 each) so the XDL
        // pipeline is not serialized on a single D->C dependency chain.
        const v16bf* bp = (const v16bf*)(&ldsbuf[cur * 1024]);
        v8f acc0 = {0.f, 0.f, 0.f, 0.f, 0.f, 0.f, 0.f, 0.f};
        v8f acc1 = acc0, acc2 = acc0;
#pragma unroll
        for (int k = 0; k < 8; ++k) {
            v16bf bhi = bp[k * 64 + lane];
            v16bf blo = bp[k * 64 + 32 + lane];
            acc0 = wmma_bf16(Ahi[k], bhi, acc0);
            acc1 = wmma_bf16(Alo[k], bhi, acc1);
            acc2 = wmma_bf16(Ahi[k], blo, acc2);
        }

        // dist = ||c||^2 - 2 * (x . c)   (||x||^2 is constant per row)
        const int code = c * 16 + (lane & 15);
        const float cn = cnorm2[code];
#pragma unroll
        for (int j = 0; j < 8; ++j) {
            float d = cn - 2.0f * (acc0[j] + (acc1[j] + acc2[j]));
            if (d < minv[j]) { minv[j] = d; mini[j] = code; }
        }

#if HAVE_ASYNC_LDS
        if (more) wait_async0();
#else
        if (more) {
            int ob = ((c + 1) & 1) * 1024;
            ldsbuf[ob + t * 4 + 0] = r0;
            ldsbuf[ob + t * 4 + 1] = r1;
            ldsbuf[ob + t * 4 + 2] = r2;
            ldsbuf[ob + t * 4 + 3] = r3;
        }
#endif
        __syncthreads();
    }

    // ---- Cross-lane argmin within each 16-lane half.
    // VGPR j: lanes 0-15 hold row j, lanes 16-31 hold row j+8.
#pragma unroll
    for (int j = 0; j < 8; ++j) {
        float v = minv[j];
        int   i = mini[j];
#pragma unroll
        for (int off = 8; off > 0; off >>= 1) {
            float ov = __shfl_xor(v, off, 16);
            int   oi = __shfl_xor(i, off, 16);
            if (ov < v || (ov == v && oi < i)) { v = ov; i = oi; }
        }
        if ((lane & 15) == 0) {
            int row = rowBase + j + ((lane >> 4) << 3);
            indices[row] = i;
        }
    }
}

// ---------------------------------------------------------------------------
// Kernel 3: finalize. One wave per row:
//   out = x + (||x - cb[idx]|| / ||rand|| + eps) * rand
// ---------------------------------------------------------------------------
__global__ void __launch_bounds__(256)
vq_finalize(const float* __restrict__ x,
            const float* __restrict__ cb,
            const float* __restrict__ rv,
            const int* __restrict__ idx,
            float* __restrict__ out) {
    const int lane = threadIdx.x & 31;
    const int wave = threadIdx.x >> 5;
    const size_t r = (size_t)blockIdx.x * 8 + wave;

    const float* xr = x + r * DD;
    const float* rr = rv + r * DD;
    const float* br = cb + (size_t)idx[r] * DD;

    float xs[8], rs[8];
    float sb = 0.f, sr = 0.f;
#pragma unroll
    for (int i = 0; i < 8; ++i) {
        int f = i * 32 + lane;
        float xv = xr[f];
        float bv = br[f];
        float rw = rr[f];
        xs[i] = xv;
        rs[i] = rw;
        float d = xv - bv;
        sb += d * d;
        sr += rw * rw;
    }
#pragma unroll
    for (int m = 16; m > 0; m >>= 1) {
        sb += __shfl_xor(sb, m, 32);
        sr += __shfl_xor(sr, m, 32);
    }
    float scale = sqrtf(sb) / sqrtf(sr) + 1e-12f;
    float* orow = out + r * DD;
#pragma unroll
    for (int i = 0; i < 8; ++i) {
        orow[i * 32 + lane] = xs[i] + scale * rs[i];
    }
}

// ---------------------------------------------------------------------------
extern "C" void kernel_launch(void* const* d_in, const int* in_sizes, int n_in,
                              void* d_out, int out_size, void* d_ws, size_t ws_size,
                              hipStream_t stream) {
    const float* x  = (const float*)d_in[0];
    const float* cb = (const float*)d_in[1];
    const float* rv = (const float*)d_in[2];
    float* out = (float*)d_out;

    // Workspace layout:
    //   [0, 4MB)            packed bf16 hi/lo codebook (KK*DD*2 bf16)
    //   [4MB, 4MB+16KB)     cnorm2 (KK floats)
    //   [4MB+16KB, +256KB)  argmin indices (NN ints)
    unsigned short* packB = (unsigned short*)d_ws;
    float* cnorm2 = (float*)((char*)d_ws + (size_t)4 * 1024 * 1024);
    int* indices  = (int*)((char*)d_ws + (size_t)4 * 1024 * 1024 + 16384);

    vq_pack_codebook<<<KK, DD, 0, stream>>>(cb, packB, cnorm2);
    vq_argmin<<<NN / 128, 256, 0, stream>>>(x, packB, cnorm2, indices);
    vq_finalize<<<NN / 8, 256, 0, stream>>>(x, cb, rv, indices, out);
}
